// MambaVAE_32804960207221
// MI455X (gfx1250) — matrix operand order repaired and checked by reading, not compiled, not run on hardware
//
#include <hip/hip_runtime.h>
#include <hip/hip_bf16.h>
#include <math.h>

// ---------------- model dims ----------------
#define DM   768
#define II   1536
#define NN   16
#define RR   48
#define KW   4
#define VV   50280
#define LE_  12
#define LD_  24
#define BZ   2
#define LL   128
#define ZZ   128
#define TT   (BZ*LL)      // 256 tokens
#define INJL 11
#define XP   (RR + 2*NN)  // 80

// ---------------- WMMA types ----------------
typedef __attribute__((ext_vector_type(16))) __bf16 v16bf;
typedef __attribute__((ext_vector_type(8)))  float  v8f;

union BFVec { v16bf v; unsigned short u[16]; };

__device__ __forceinline__ unsigned short f2bf(float x) {
  unsigned int b = __float_as_uint(x);
  unsigned int r = (b + 0x7FFFu + ((b >> 16) & 1u)) >> 16;   // RNE
  return (unsigned short)r;
}

__device__ __forceinline__ float siluf(float v)     { return v / (1.f + __expf(-v)); }
__device__ __forceinline__ float softplusf(float v) { return v > 20.f ? v : log1pf(__expf(v)); }

// =====================================================================
// Generic WMMA GEMM:  C[M,N] = A[M,K] @ W[N,K]^T  (+bias[N]) (+res[M,N])
// block = 128 threads = 4 waves; each wave owns TWO adjacent 16x16 C
// tiles (1x2 N-tiling): the A fragment is loaded/converted once per
// 32-wide K chunk and feeds two independent back-to-back WMMAs.
// Interior tiles: unguarded float4 loads (global_load_b128).
// Edge tiles / K tails: clamped unconditional loads + v_cndmask zeroing.
// =====================================================================
__global__ void k_gemm_bf16wmma(const float* __restrict__ A,
                                const float* __restrict__ W,
                                const float* __restrict__ bias,
                                const float* __restrict__ res,
                                float* __restrict__ C,
                                int M, int N, int Kd,
                                int lda, int ldw, int ldc)
{
  const int lane = threadIdx.x & 31;
  const int wv   = threadIdx.x >> 5;
  const int lo   = lane & 15;
  const int hi   = lane >> 4;
  const int nb   = (blockIdx.x * 8 + wv * 2) * 16;   // first of 2 N tiles
  const int mb   = blockIdx.y * 16;
  if (nb >= N) return;                               // wave-uniform exit

  const int m  = mb + lo;
  const bool mok = (m < M);
  const float* Arow = A + (size_t)(mok ? m : (M - 1)) * lda;

  const int n0 = nb + lo;
  const int n1 = nb + 16 + lo;
  const bool n0ok = (n0 < N);
  const bool n1ok = (n1 < N);
  const float* Wrow0 = W + (size_t)(n0ok ? n0 : (N - 1)) * ldw;
  const float* Wrow1 = W + (size_t)(n1ok ? n1 : (N - 1)) * ldw;

  v8f acc0 = {0.f, 0.f, 0.f, 0.f, 0.f, 0.f, 0.f, 0.f};
  v8f acc1 = {0.f, 0.f, 0.f, 0.f, 0.f, 0.f, 0.f, 0.f};

  // branchless guarded fragment loaders (edge tiles, K tails)
  auto loadA_g = [&](int kb, BFVec& af) {
#pragma unroll
    for (int i = 0; i < 16; ++i) {
      int ka  = kb + hi * 16 + i;                          // A 16x32 layout (contiguous k)
      int kac = ka < Kd ? ka : Kd - 1;
      float av = Arow[kac];
      af.u[i] = f2bf((mok && ka < Kd) ? av : 0.f);
    }
  };
  auto loadB_g = [&](int kb, const float* Wrow, bool nok, BFVec& bf) {
#pragma unroll
    for (int i = 0; i < 16; ++i) {
      int kbb = kb + hi * 16 + i;                          // B 32x16 layout
      int kbc = kbb < Kd ? kbb : Kd - 1;
      float bv = Wrow[kbc];
      bf.u[i] = f2bf((nok && kbb < Kd) ? bv : 0.f);
    }
  };
  auto kchunk_guarded = [&](int kb) {
    BFVec af, bf0, bf1;
    loadA_g(kb, af);
    loadB_g(kb, Wrow0, n0ok, bf0);
    loadB_g(kb, Wrow1, n1ok, bf1);
    acc0 = __builtin_amdgcn_wmma_f32_16x16x32_bf16(
        false, af.v, false, bf0.v, (short)0, acc0, false, false);
    acc1 = __builtin_amdgcn_wmma_f32_16x16x32_bf16(
        false, af.v, false, bf1.v, (short)0, acc1, false, false);
  };

  const int kmain = Kd & ~31;
  if ((mb + 16 <= M) && (nb + 32 <= N)) {
    // ---------------- fast interior path ----------------
    for (int kb = 0; kb < kmain; kb += 32) {
      if (kb + 32 < kmain) {
        __builtin_prefetch(Arow  + kb + 32 + hi * 16, 0, 1);
        __builtin_prefetch(Wrow0 + kb + 32 + hi * 16, 0, 1);
        __builtin_prefetch(Wrow1 + kb + 32 + hi * 16, 0, 1);
      }
      const float4* ap = (const float4*)(Arow + kb + hi * 16);
      float4 a0 = ap[0];        // k = kb+hi*16 + 0..3
      float4 a1 = ap[1];        //              + 4..7
      float4 a2 = ap[2];        //              + 8..11
      float4 a3 = ap[3];        //              + 12..15
      const float4* b0p = (const float4*)(Wrow0 + kb + hi * 16);
      float4 p0 = b0p[0], p1 = b0p[1], p2 = b0p[2], p3 = b0p[3];
      const float4* b1p = (const float4*)(Wrow1 + kb + hi * 16);
      float4 q0 = b1p[0], q1 = b1p[1], q2 = b1p[2], q3 = b1p[3];

      BFVec af, bf0, bf1;
      af.u[0]  = f2bf(a0.x); af.u[1]  = f2bf(a0.y); af.u[2]  = f2bf(a0.z); af.u[3]  = f2bf(a0.w);
      af.u[4]  = f2bf(a1.x); af.u[5]  = f2bf(a1.y); af.u[6]  = f2bf(a1.z); af.u[7]  = f2bf(a1.w);
      af.u[8]  = f2bf(a2.x); af.u[9]  = f2bf(a2.y); af.u[10] = f2bf(a2.z); af.u[11] = f2bf(a2.w);
      af.u[12] = f2bf(a3.x); af.u[13] = f2bf(a3.y); af.u[14] = f2bf(a3.z); af.u[15] = f2bf(a3.w);
      bf0.u[0]  = f2bf(p0.x); bf0.u[1]  = f2bf(p0.y); bf0.u[2]  = f2bf(p0.z); bf0.u[3]  = f2bf(p0.w);
      bf0.u[4]  = f2bf(p1.x); bf0.u[5]  = f2bf(p1.y); bf0.u[6]  = f2bf(p1.z); bf0.u[7]  = f2bf(p1.w);
      bf0.u[8]  = f2bf(p2.x); bf0.u[9]  = f2bf(p2.y); bf0.u[10] = f2bf(p2.z); bf0.u[11] = f2bf(p2.w);
      bf0.u[12] = f2bf(p3.x); bf0.u[13] = f2bf(p3.y); bf0.u[14] = f2bf(p3.z); bf0.u[15] = f2bf(p3.w);
      bf1.u[0]  = f2bf(q0.x); bf1.u[1]  = f2bf(q0.y); bf1.u[2]  = f2bf(q0.z); bf1.u[3]  = f2bf(q0.w);
      bf1.u[4]  = f2bf(q1.x); bf1.u[5]  = f2bf(q1.y); bf1.u[6]  = f2bf(q1.z); bf1.u[7]  = f2bf(q1.w);
      bf1.u[8]  = f2bf(q2.x); bf1.u[9]  = f2bf(q2.y); bf1.u[10] = f2bf(q2.z); bf1.u[11] = f2bf(q2.w);
      bf1.u[12] = f2bf(q3.x); bf1.u[13] = f2bf(q3.y); bf1.u[14] = f2bf(q3.z); bf1.u[15] = f2bf(q3.w);

      acc0 = __builtin_amdgcn_wmma_f32_16x16x32_bf16(
          false, af.v, false, bf0.v, (short)0, acc0, false, false);
      acc1 = __builtin_amdgcn_wmma_f32_16x16x32_bf16(
          false, af.v, false, bf1.v, (short)0, acc1, false, false);
    }
    if (kmain < Kd) kchunk_guarded(kmain);
  } else {
    // ---------------- edge path ----------------
    for (int kb = 0; kb < Kd; kb += 32) kchunk_guarded(kb);
  }

  // ---------------- epilogue (both tiles) ----------------
  float bv0 = (bias != nullptr && n0ok) ? bias[n0] : 0.f;
  float bv1 = (bias != nullptr && n1ok) ? bias[n1] : 0.f;
#pragma unroll
  for (int r = 0; r < 8; ++r) {
    int mm = mb + hi * 8 + r;                // C/D layout: M = r + hi*8, N = lo
    if (mm < M) {
      if (n0ok) {
        float o = acc0[r] + bv0;
        if (res != nullptr) o += res[(size_t)mm * ldc + n0];
        C[(size_t)mm * ldc + n0] = o;
      }
      if (n1ok) {
        float o = acc1[r] + bv1;
        if (res != nullptr) o += res[(size_t)mm * ldc + n1];
        C[(size_t)mm * ldc + n1] = o;
      }
    }
  }
}

// ---------------- mask / loss-mask selection ----------------
__global__ void k_masks(const int* __restrict__ fullm, const int* __restrict__ qm,
                        const int* __restrict__ flossm, const int* __restrict__ idxm,
                        float* __restrict__ maskA, float* __restrict__ lm)
{
  int t = blockIdx.x * blockDim.x + threadIdx.x;
  if (t >= TT) return;
  int b = t / LL;
  bool sel = idxm[b] > 0;
  maskA[t] = sel ? (float)qm[t] : (float)fullm[t];
  lm[t]    = sel ? (float)qm[t] : (float)flossm[t];
}

__global__ void k_zero_acc(float* __restrict__ acc)
{
  if (threadIdx.x < 4) acc[threadIdx.x] = 0.f;
}

// ---------------- embedding gather ----------------
__global__ void k_embed(const float* __restrict__ E, const int* __restrict__ ids,
                        float* __restrict__ h)
{
  int idx = blockIdx.x * blockDim.x + threadIdx.x;
  if (idx >= TT * DM) return;
  int t = idx / DM, d = idx % DM;
  h[idx] = E[(size_t)ids[t] * DM + d];
}

// ---------------- rmsnorm (one block per token) ----------------
__global__ void k_rmsnorm(const float* __restrict__ x, const float* __restrict__ w,
                          float* __restrict__ o)
{
  __shared__ float red[256];
  int t = blockIdx.x;
  const float* xr = x + (size_t)t * DM;
  float s = 0.f;
  for (int d = threadIdx.x; d < DM; d += 256) { float v = xr[d]; s += v * v; }
  red[threadIdx.x] = s;
  __syncthreads();
  for (int st = 128; st > 0; st >>= 1) {
    if (threadIdx.x < st) red[threadIdx.x] += red[threadIdx.x + st];
    __syncthreads();
  }
  float inv = rsqrtf(red[0] / (float)DM + 1e-5f);
  for (int d = threadIdx.x; d < DM; d += 256)
    o[(size_t)t * DM + d] = xr[d] * inv * w[d];
}

// ---------------- depthwise causal conv (K=4) + SiLU + mask ----------------
__global__ void k_conv(const float* __restrict__ xz, const float* __restrict__ cw,
                       const float* __restrict__ cb, const float* __restrict__ mask,
                       float* __restrict__ xsb)
{
  int idx = blockIdx.x * blockDim.x + threadIdx.x;
  if (idx >= TT * II) return;
  int t = idx / II, c = idx % II;
  int b = t / LL, l = t % LL;
  float acc = 0.f;
#pragma unroll
  for (int k = 0; k < KW; ++k) {
    int ls = l - (KW - 1) + k;
    if (ls >= 0) {
      int ts = b * LL + ls;
      acc += cw[c * KW + k] * xz[(size_t)ts * (2 * II) + c] * mask[ts];
    }
  }
  xsb[idx] = siluf(acc + cb[c]) * mask[t];
}

// ---------------- selective scan (one thread per (b, channel)) ----------------
__global__ void k_scan(const float* __restrict__ xz,   const float* __restrict__ xsb,
                       const float* __restrict__ dtmm, const float* __restrict__ proj,
                       const float* __restrict__ dtb,  const float* __restrict__ Alog,
                       const float* __restrict__ Dp,   const float* __restrict__ h0,
                       float* __restrict__ g,          float* __restrict__ hT)
{
  int idx = blockIdx.x * blockDim.x + threadIdx.x;
  if (idx >= BZ * II) return;
  int b = idx / II, i = idx % II;

  float Ai[NN], st[NN];
#pragma unroll
  for (int n = 0; n < NN; ++n) {
    Ai[n] = -__expf(Alog[i * NN + n]);
    st[n] = (h0 != nullptr) ? h0[(size_t)b * II * NN + (size_t)i * NN + n] : 0.f;
  }
  float bia = dtb[i], Dv = Dp[i];

  for (int l = 0; l < LL; ++l) {
    int t = b * LL + l;
    float dt = softplusf(dtmm[(size_t)t * II + i] + bia);
    float xv = xsb[(size_t)t * II + i];
    const float* pr = proj + (size_t)t * XP;
    float y = 0.f;
#pragma unroll
    for (int n = 0; n < NN; ++n) {
      float dA = __expf(dt * Ai[n]);
      st[n] = dA * st[n] + dt * pr[RR + n] * xv;
      y += st[n] * pr[RR + NN + n];
    }
    y += Dv * xv;
    float z = xz[(size_t)t * (2 * II) + II + i];
    g[(size_t)t * II + i] = siluf(z) * y;
  }
#pragma unroll
  for (int n = 0; n < NN; ++n)
    hT[(size_t)b * II * NN + (size_t)i * NN + n] = st[n];
}

// ---------------- hT (B,I,N) -> s (B*N, I) ----------------
__global__ void k_transpose_hT(const float* __restrict__ hT, float* __restrict__ sbuf)
{
  int idx = blockIdx.x * blockDim.x + threadIdx.x;
  if (idx >= BZ * NN * II) return;
  int r = idx / II, i = idx % II;
  int b = r / NN, n = r % NN;
  sbuf[(size_t)r * II + i] = hT[(size_t)b * II * NN + (size_t)i * NN + n];
}

// ---------------- VAE sample + KL term ----------------
__global__ void k_states(const float* __restrict__ mu, const float* __restrict__ lv,
                         const float* __restrict__ noise, float* __restrict__ stbuf,
                         float* __restrict__ out_states, float* __restrict__ acc)
{
  int idx = blockIdx.x * blockDim.x + threadIdx.x;
  if (idx >= BZ * NN * ZZ) return;
  float m = mu[idx], l = lv[idx];
  float s = fminf(fmaxf(m, -1.f), 1.f) + noise[idx] * __expf(0.5f * l);
  stbuf[idx] = s;
  out_states[idx] = s;
  float kt = fmaxf(-0.5f * (1.f + l - m * m - __expf(l)), 0.05f);
  atomicAdd(&acc[0], kt);
}

// ---------------- inj rows (B*N, I) -> h0 (B,I,N) ----------------
__global__ void k_transpose_inj(const float* __restrict__ injrow, float* __restrict__ h0)
{
  int idx = blockIdx.x * blockDim.x + threadIdx.x;
  if (idx >= BZ * NN * II) return;
  int r = idx / II, i = idx % II;
  int b = r / NN, n = r % NN;
  h0[(size_t)b * II * NN + (size_t)i * NN + n] = injrow[(size_t)r * II + i];
}

// ---------------- log-softmax cross-entropy (one block per (b, l<L-1)) ------
__global__ void k_ce(const float* __restrict__ logits, const int* __restrict__ ids,
                     const float* __restrict__ lm, float* __restrict__ acc)
{
  __shared__ float red[256];
  int r = blockIdx.x;                 // 0 .. BZ*(LL-1)-1
  int b = r / (LL - 1), l = r % (LL - 1);
  const float* row = logits + (size_t)(b * LL + l) * VV;

  float mx = -3.4e38f;
  for (int v = threadIdx.x; v < VV; v += 256) mx = fmaxf(mx, row[v]);
  red[threadIdx.x] = mx;
  __syncthreads();
  for (int s = 128; s > 0; s >>= 1) {
    if (threadIdx.x < s) red[threadIdx.x] = fmaxf(red[threadIdx.x], red[threadIdx.x + s]);
    __syncthreads();
  }
  mx = red[0];
  __syncthreads();

  float se = 0.f;
  for (int v = threadIdx.x; v < VV; v += 256) se += __expf(row[v] - mx);
  red[threadIdx.x] = se;
  __syncthreads();
  for (int s = 128; s > 0; s >>= 1) {
    if (threadIdx.x < s) red[threadIdx.x] += red[threadIdx.x + s];
    __syncthreads();
  }

  if (threadIdx.x == 0) {
    int tgt = ids[b * LL + l + 1];
    float lmv = lm[b * LL + l + 1];
    float ce = -(row[tgt] - mx - logf(red[0]));
    atomicAdd(&acc[1], ce * lmv);
    atomicAdd(&acc[2], lmv);
  }
}

__global__ void k_finalize(const float* __restrict__ acc, float* __restrict__ out)
{
  if (threadIdx.x == 0) {
    out[0] = acc[1] / fmaxf(acc[2], 1.f);  // loss
    out[1] = acc[0] / (float)BZ;           // kl
  }
}

// =====================================================================
// host driver
// =====================================================================
extern "C" void kernel_launch(void* const* d_in, const int* in_sizes, int n_in,
                              void* d_out, int out_size, void* d_ws, size_t ws_size,
                              hipStream_t stream)
{
  (void)in_sizes; (void)n_in; (void)out_size; (void)ws_size;

  const float* E          = (const float*)d_in[0];
  const float* enc_norm   = (const float*)d_in[1];
  const float* enc_in_w   = (const float*)d_in[2];
  const float* enc_conv_w = (const float*)d_in[3];
  const float* enc_conv_b = (const float*)d_in[4];
  const float* enc_xproj  = (const float*)d_in[5];
  const float* enc_dt_w   = (const float*)d_in[6];
  const float* enc_dt_b   = (const float*)d_in[7];
  const float* enc_A_log  = (const float*)d_in[8];
  const float* enc_D      = (const float*)d_in[9];
  const float* enc_out_w  = (const float*)d_in[10];
  const float* dec_norm   = (const float*)d_in[11];
  const float* dec_in_w   = (const float*)d_in[12];
  const float* dec_conv_w = (const float*)d_in[13];
  const float* dec_conv_b = (const float*)d_in[14];
  const float* dec_xproj  = (const float*)d_in[15];
  const float* dec_dt_w   = (const float*)d_in[16];
  const float* dec_dt_b   = (const float*)d_in[17];
  const float* dec_A_log  = (const float*)d_in[18];
  const float* dec_D      = (const float*)d_in[19];
  const float* dec_out_w  = (const float*)d_in[20];
  const float* norm_f     = (const float*)d_in[21];
  const float* mean_w     = (const float*)d_in[22];
  const float* mean_b     = (const float*)d_in[23];
  const float* lv_w       = (const float*)d_in[24];
  const float* lv_b       = (const float*)d_in[25];
  const float* st_w       = (const float*)d_in[26];
  const float* st_b       = (const float*)d_in[27];
  const float* noise      = (const float*)d_in[28];
  const int*   ids        = (const int*)d_in[29];
  const int*   full_mask  = (const int*)d_in[30];
  const int*   q_mask     = (const int*)d_in[31];
  const int*   floss_mask = (const int*)d_in[32];
  const int*   idx_mask   = (const int*)d_in[33];

  float* out = (float*)d_out;
  float* out_logits = out + 2;
  float* out_states = out + 2 + (size_t)BZ * LL * VV;

  // ---- workspace carve-up (floats; all 16B aligned) ----
  float* p = (float*)d_ws;
  float* maskA  = p; p += 256;
  float* lm     = p; p += 256;
  float* acc    = p; p += 512;            // [kl, loss_num, loss_den, pad]
  float* h1     = p; p += (size_t)TT * DM;
  float* h2     = p; p += (size_t)TT * DM;
  float* xn     = p; p += (size_t)TT * DM;
  float* xz     = p; p += (size_t)TT * 2 * II;
  float* xsb    = p; p += (size_t)TT * II;
  float* proj   = p; p += (size_t)TT * XP;
  float* dtmm   = p; p += (size_t)TT * II;
  float* gbuf   = p; p += (size_t)TT * II;
  float* hTbuf  = p; p += (size_t)BZ * II * NN;
  float* sbuf   = p; p += (size_t)BZ * NN * II;
  float* mu     = p; p += (size_t)BZ * NN * ZZ;
  float* lv     = p; p += (size_t)BZ * NN * ZZ;
  float* stbuf  = p; p += (size_t)BZ * NN * ZZ;
  float* injrow = p; p += (size_t)BZ * NN * II;
  float* h0dec  = p; p += (size_t)BZ * II * NN;

  auto gemm = [&](const float* A, const float* W, const float* bias, const float* res,
                  float* C, int M, int N, int Kd, int lda, int ldw, int ldc) {
    int ntiles = (N + 15) / 16;
    dim3 grid((unsigned)((ntiles + 7) / 8), (unsigned)((M + 15) / 16));
    k_gemm_bf16wmma<<<grid, dim3(128), 0, stream>>>(A, W, bias, res, C,
                                                    M, N, Kd, lda, ldw, ldc);
  };

  auto mixer = [&](const float* nw, const float* iw, const float* cw, const float* cb,
                   const float* xpw, const float* dtw, const float* dtb,
                   const float* Alog, const float* Dp, const float* ow,
                   const float* h0, float* hcur, float* hnext) {
    k_rmsnorm<<<TT, 256, 0, stream>>>(hcur, nw, xn);
    gemm(xn, iw, nullptr, nullptr, xz, TT, 2 * II, DM, DM, DM, 2 * II);
    k_conv<<<(TT * II + 255) / 256, 256, 0, stream>>>(xz, cw, cb, maskA, xsb);
    gemm(xsb, xpw, nullptr, nullptr, proj, TT, XP, II, II, II, XP);
    gemm(proj, dtw, nullptr, nullptr, dtmm, TT, II, RR, XP, RR, II); // K=48 -> tail chunk
    k_scan<<<(BZ * II + 127) / 128, 128, 0, stream>>>(xz, xsb, dtmm, proj,
                                                      dtb, Alog, Dp, h0, gbuf, hTbuf);
    gemm(gbuf, ow, nullptr, hcur, hnext, TT, DM, II, II, II, DM); // residual fused
  };

  // ---- prologue ----
  k_masks<<<1, 256, 0, stream>>>(full_mask, q_mask, floss_mask, idx_mask, maskA, lm);
  k_zero_acc<<<1, 32, 0, stream>>>(acc);

  // ---- encoder ----
  k_embed<<<(TT * DM + 255) / 256, 256, 0, stream>>>(E, ids, h1);
  float* hc = h1; float* hn = h2;
  for (int li = 0; li < LE_; ++li) {
    mixer(enc_norm   + (size_t)li * DM,
          enc_in_w   + (size_t)li * 2 * II * DM,
          enc_conv_w + (size_t)li * II * KW,
          enc_conv_b + (size_t)li * II,
          enc_xproj  + (size_t)li * XP * II,
          enc_dt_w   + (size_t)li * II * RR,
          enc_dt_b   + (size_t)li * II,
          enc_A_log  + (size_t)li * II * NN,
          enc_D      + (size_t)li * II,
          enc_out_w  + (size_t)li * DM * II,
          nullptr, hc, hn);
    float* tmp = hc; hc = hn; hn = tmp;
  }
  // hTbuf now holds ssm from the last encoder layer

  // ---- VAE head ----
  k_transpose_hT<<<(BZ * NN * II + 255) / 256, 256, 0, stream>>>(hTbuf, sbuf);
  gemm(sbuf, mean_w, mean_b, nullptr, mu, BZ * NN, ZZ, II, II, II, ZZ);
  gemm(sbuf, lv_w,   lv_b,   nullptr, lv, BZ * NN, ZZ, II, II, II, ZZ);
  k_states<<<(BZ * NN * ZZ + 255) / 256, 256, 0, stream>>>(mu, lv, noise,
                                                           stbuf, out_states, acc);
  gemm(stbuf, st_w, st_b, nullptr, injrow, BZ * NN, II, ZZ, ZZ, ZZ, II);
  k_transpose_inj<<<(BZ * NN * II + 255) / 256, 256, 0, stream>>>(injrow, h0dec);

  // ---- decoder ----
  k_embed<<<(TT * DM + 255) / 256, 256, 0, stream>>>(E, ids, h1);
  hc = h1; hn = h2;
  for (int li = 0; li < LD_; ++li) {
    mixer(dec_norm   + (size_t)li * DM,
          dec_in_w   + (size_t)li * 2 * II * DM,
          dec_conv_w + (size_t)li * II * KW,
          dec_conv_b + (size_t)li * II,
          dec_xproj  + (size_t)li * XP * II,
          dec_dt_w   + (size_t)li * II * RR,
          dec_dt_b   + (size_t)li * II,
          dec_A_log  + (size_t)li * II * NN,
          dec_D      + (size_t)li * II,
          dec_out_w  + (size_t)li * DM * II,
          (li == INJL) ? h0dec : nullptr, hc, hn);
    float* tmp = hc; hc = hn; hn = tmp;
  }

  // ---- logits + loss ----
  k_rmsnorm<<<TT, 256, 0, stream>>>(hc, norm_f, xn);
  gemm(xn, E, nullptr, nullptr, out_logits, TT, VV, DM, DM, DM, VV);
  k_ce<<<BZ * (LL - 1), 256, 0, stream>>>(out_logits, ids, lm, acc);
  k_finalize<<<1, 32, 0, stream>>>(acc, out);
}
